// MultiHeadAttention_38225208935009
// MI455X (gfx1250) — matrix operand order repaired and checked by reading, not compiled, not run on hardware
//
#include <hip/hip_runtime.h>

// ---------------------------------------------------------------------------
// MultiHeadAttention for gfx1250 (MI455X): B=2, S=2048, D=1024, H=16, DK=64
// All matmuls via V_WMMA_F32_16X16X32_BF16 (bf16 A/B, f32 accumulate).
// K-tile / bf16 X-tile staging via GLOBAL_LOAD_ASYNC_TO_LDS_B128 (ASYNCcnt).
// fp32->bf16 staging uses packed vector converts (v_cvt_pk_bf16_f32).
// ---------------------------------------------------------------------------

typedef __attribute__((ext_vector_type(16))) __bf16 v16bf;
typedef __attribute__((ext_vector_type(8)))  float  v8f;
typedef __attribute__((ext_vector_type(2)))  float  v2f;
typedef __attribute__((ext_vector_type(2)))  __bf16 v2bf;

#define S_LEN 2048
#define D_DIM 1024
#define NHEAD 16
#define DKDIM 64

// Packed fp32x2 -> bf16x2 (lowers to v_cvt_pk_bf16_f32).
__device__ __forceinline__ unsigned pack_bf2(float a, float b) {
  v2f f;
  f.x = a;
  f.y = b;
  union { v2bf h; unsigned u; } t;
  t.h = __builtin_convertvector(f, v2bf);
  return t.u;
}

// Load a 16x32 bf16 WMMA fragment (A or B layout) as two 16-byte runs.
__device__ __forceinline__ v16bf ld_frag(const __bf16* p0, const __bf16* p1) {
  union { uint4 q[2]; v16bf v; } f;
  f.q[0] = *(const uint4*)p0;
  f.q[1] = *(const uint4*)p1;
  return f.v;
}

__device__ __forceinline__ v8f wmma_bf16(v16bf a, v16bf b, v8f c) {
  return __builtin_amdgcn_wmma_f32_16x16x32_bf16(
      /*neg_a=*/false, a, /*neg_b=*/false, b,
      /*c_mod=*/(short)0, c, /*reuse_a=*/false, /*reuse_b=*/false);
}

// Async 16-byte copy global -> LDS (GV mode, tracked by ASYNCcnt).
// LDS byte offset = low 32 bits of the generic __shared__ pointer
// (aperture rule: LDS address is the pointer's low 32 bits).
__device__ __forceinline__ void async_ld_b128(void* lds, const void* gptr) {
  const unsigned lds_off = (unsigned)(unsigned long long)lds;
  asm volatile("global_load_async_to_lds_b128 %0, %1, off"
               : : "v"(lds_off), "v"(gptr) : "memory");
}

__device__ __forceinline__ void wait_asynccnt0() {
  asm volatile("s_wait_asynccnt 0x0" : : : "memory");
}

// ---------------------------------------------------------------------------
// GEMM core: Y[M,N] = X[M,1024] @ W[N,1024]^T + bias (tile 128x128, K-chunk 32)
// MODE 0: X fp32 -> bf16 head-split [B,H,S,DK] output.
// MODE 1: X bf16 (async-staged) -> fp32 flat [M,1024] output.
// 8 waves: (wid&3) -> 32-row band, (wid>>2) -> 64-col band; 2x4 WMMA frags.
// ---------------------------------------------------------------------------
template <int MODE>
__device__ __forceinline__ void gemm_core(const float* Xf, const __bf16* Xb,
                                          const float* W, const float* bias,
                                          __bf16* Yb, float* Yf) {
  __shared__ __bf16 Xs[128 * 40];  // 128 rows x 32 K (stride 40, 16B-aligned)
  __shared__ __bf16 Ws[128 * 40];  // 128 N-rows x 32 K

  const int tid   = threadIdx.x;
  const int wid   = tid >> 5;
  const int lane  = tid & 31;
  const int lhalf = lane >> 4;
  const int l16   = lane & 15;
  const int m0 = blockIdx.x * 128;
  const int n0 = blockIdx.y * 128;
  const int wm = (wid & 3) * 32;
  const int wn = (wid >> 2) * 64;

  v8f acc[2][4] = {};

  for (int k0 = 0; k0 < 1024; k0 += 32) {
    __syncthreads();
    if (MODE == 0) {
      // stage X chunk 128x32 fp32 -> bf16
#pragma unroll
      for (int rep = 0; rep < 4; ++rep) {
        const int r = rep * 32 + (tid >> 3);
        const int c = (tid & 7) * 4;
        const float4 x = *(const float4*)(Xf + (size_t)(m0 + r) * 1024 + k0 + c);
        uint2 p;
        p.x = pack_bf2(x.x, x.y);
        p.y = pack_bf2(x.z, x.w);
        *(uint2*)&Xs[r * 40 + c] = p;
      }
    } else {
      // stage X chunk 128x32 bf16 via async copy to LDS
#pragma unroll
      for (int rep = 0; rep < 2; ++rep) {
        const int r = rep * 64 + (tid >> 2);
        const int c = (tid & 3) * 8;
        async_ld_b128(&Xs[r * 40 + c],
                      Xb + (size_t)(m0 + r) * 1024 + k0 + c);
      }
    }
    // stage W chunk 128x32 fp32 -> bf16 (rows of W contiguous in K)
#pragma unroll
    for (int rep = 0; rep < 4; ++rep) {
      const int r = rep * 32 + (tid >> 3);
      const int c = (tid & 7) * 4;
      const float4 w = *(const float4*)(W + (size_t)(n0 + r) * 1024 + k0 + c);
      uint2 p;
      p.x = pack_bf2(w.x, w.y);
      p.y = pack_bf2(w.z, w.w);
      *(uint2*)&Ws[r * 40 + c] = p;
    }
    if (MODE == 1) wait_asynccnt0();
    __syncthreads();

    v16bf a[2], b[4];
#pragma unroll
    for (int mi = 0; mi < 2; ++mi) {
      const __bf16* xr = &Xs[(wm + mi * 16 + l16) * 40];
      a[mi] = ld_frag(xr + lhalf * 8, xr + 16 + lhalf * 8);  // A layout
    }
#pragma unroll
    for (int ni = 0; ni < 4; ++ni) {
      const __bf16* wr = &Ws[(wn + ni * 16 + l16) * 40];
      b[ni] = ld_frag(wr + lhalf * 8, wr + 16 + lhalf * 8);  // B layout: same K order as A
    }
#pragma unroll
    for (int mi = 0; mi < 2; ++mi)
#pragma unroll
      for (int ni = 0; ni < 4; ++ni)
        acc[mi][ni] = wmma_bf16(a[mi], b[ni], acc[mi][ni]);
  }

  // epilogue: bias + store
#pragma unroll
  for (int ni = 0; ni < 4; ++ni) {
    const int n = n0 + wn + ni * 16 + l16;
    const float bv = bias[n];
#pragma unroll
    for (int mi = 0; mi < 2; ++mi) {
#pragma unroll
      for (int r = 0; r < 8; ++r) {
        const int m = m0 + wm + mi * 16 + lhalf * 8 + r;  // C layout row
        const float v = acc[mi][ni][r] + bv;
        if (MODE == 0) {
          const int bb = m >> 11, sq = m & 2047;
          const int h = n >> 6, dk = n & 63;
          Yb[(((size_t)bb * NHEAD + h) * S_LEN + sq) * DKDIM + dk] = (__bf16)v;
        } else {
          Yf[(size_t)m * 1024 + n] = v;
        }
      }
    }
  }
}

__global__ __launch_bounds__(256) void qkv_proj_kernel(
    const float* q, const float* k, const float* v,
    const float* wq, const float* bq, const float* wk, const float* bk,
    const float* wv, const float* bv,
    __bf16* qh, __bf16* kh, __bf16* vh) {
  const float* X; const float* W; const float* bias; __bf16* Y;
  if (blockIdx.z == 0)      { X = q; W = wq; bias = bq; Y = qh; }
  else if (blockIdx.z == 1) { X = k; W = wk; bias = bk; Y = kh; }
  else                      { X = v; W = wv; bias = bv; Y = vh; }
  gemm_core<0>(X, nullptr, W, bias, Y, nullptr);
}

__global__ __launch_bounds__(256) void out_proj_kernel(
    const __bf16* ctx, const float* wo, const float* bo, float* out) {
  gemm_core<1>(nullptr, ctx, wo, bo, nullptr, out);
}

// ---------------------------------------------------------------------------
// Flash attention: one block = 128 q-rows of one (b,h); 8 waves x 16 rows.
// K chunk async-staged to LDS; V chunk transposed in LDS. Online softmax.
// ---------------------------------------------------------------------------
__global__ __launch_bounds__(256) void attn_kernel(
    const __bf16* Qh, const __bf16* Kh, const __bf16* Vh, __bf16* Ctx) {
  __shared__ __bf16 Ks[64 * 72];      // [k'][d]   (stride 72, 16B-aligned)
  __shared__ __bf16 Vts[64 * 72];     // [d][k']   transposed
  __shared__ __bf16 Ps[8 * 16 * 72];  // per-wave P tiles 16x64

  const int tid   = threadIdx.x;
  const int wid   = tid >> 5;
  const int lane  = tid & 31;
  const int lhalf = lane >> 4;
  const int l16   = lane & 15;
  const int bh = blockIdx.y;       // b*16 + h
  const int q0 = blockIdx.x * 128;
  const int qrow = q0 + wid * 16;  // this wave's first q row
  const size_t base = (size_t)bh * S_LEN * DKDIM;

  // Q fragments for this wave's 16 rows: 16x64 = two 16x32 A-frags
  v16bf qa[2];
#pragma unroll
  for (int kk = 0; kk < 2; ++kk) {
    const __bf16* qp = Qh + base + (size_t)(qrow + l16) * DKDIM + kk * 32;
    qa[kk] = ld_frag(qp + lhalf * 8, qp + 16 + lhalf * 8);
  }

  v8f ctx[4] = {};
  float mrun[8], lrun[8];
#pragma unroll
  for (int r = 0; r < 8; ++r) { mrun[r] = -1e30f; lrun[r] = 0.0f; }

  const int ncov = 2 * blockIdx.x + 2;  // causal coverage of q0..q0+127

  for (int kc = 0; kc < ncov; ++kc) {
    // prefetch next chunk's K/V rows while we work on this one
    if (kc + 1 < ncov) {
      const size_t nrow = (size_t)((kc + 1) * 64 + (tid & 63)) * DKDIM;
      __builtin_prefetch(Kh + base + nrow, 0, 0);
      __builtin_prefetch(Vh + base + nrow, 0, 0);
    }
    __syncthreads();
    // stage K chunk [64 keys x 64 d] via async copy to LDS
    {
      const int r = tid >> 2;
      const int c = (tid & 3) * 16;
      const __bf16* gsrc = Kh + base + (size_t)(kc * 64 + r) * DKDIM + c;
      async_ld_b128(&Ks[r * 72 + c], gsrc);
      async_ld_b128(&Ks[r * 72 + c + 8], gsrc + 8);
    }
    // stage V chunk transposed: Vts[d][k]
    {
      const int kq = tid & 63;
      const int ds = (tid >> 6) * 16;
      const __bf16* src = Vh + base + (size_t)(kc * 64 + kq) * DKDIM + ds;
#pragma unroll
      for (int e = 0; e < 16; ++e)
        Vts[(ds + e) * 72 + kq] = src[e];
    }
    wait_asynccnt0();
    __syncthreads();

    // waves whose rows are entirely above this key chunk skip compute
    // (wave-uniform branch: EXEC stays all-ones inside -> WMMA legal)
    if (kc * 64 <= qrow + 15) {
      // ---- scores: S = Q @ K^T, 16 x 64 over d=64 ----
      v8f s[4] = {};
#pragma unroll
      for (int nf = 0; nf < 4; ++nf) {
#pragma unroll
        for (int kk = 0; kk < 2; ++kk) {
          const __bf16* kr = &Ks[(nf * 16 + l16) * 72 + kk * 32];
          s[nf] = wmma_bf16(qa[kk], ld_frag(kr + lhalf * 8, kr + 16 + lhalf * 8), s[nf]);
        }
      }
      // ---- scale + causal mask ----
#pragma unroll
      for (int nf = 0; nf < 4; ++nf) {
        const int kg = kc * 64 + nf * 16 + l16;
#pragma unroll
        for (int r = 0; r < 8; ++r) {
          const int qg = qrow + lhalf * 8 + r;
          float val = s[nf][r] * 0.125f;  // 1/sqrt(64)
          if (kg > qg) val = -1e9f;
          s[nf][r] = val;
        }
      }
      // ---- online softmax update ----
      float mnew[8];
#pragma unroll
      for (int r = 0; r < 8; ++r) {
        float mx = fmaxf(fmaxf(s[0][r], s[1][r]), fmaxf(s[2][r], s[3][r]));
        mx = fmaxf(mx, __shfl_xor(mx, 1, 32));
        mx = fmaxf(mx, __shfl_xor(mx, 2, 32));
        mx = fmaxf(mx, __shfl_xor(mx, 4, 32));
        mx = fmaxf(mx, __shfl_xor(mx, 8, 32));
        mnew[r] = fmaxf(mrun[r], mx);
      }
#pragma unroll
      for (int r = 0; r < 8; ++r) {
        const float alpha = __expf(mrun[r] - mnew[r]);
        mrun[r] = mnew[r];
        float rs = 0.0f;
#pragma unroll
        for (int nf = 0; nf < 4; ++nf) {
          const float p = __expf(s[nf][r] - mnew[r]);
          s[nf][r] = p;
          rs += p;
        }
        rs += __shfl_xor(rs, 1, 32);
        rs += __shfl_xor(rs, 2, 32);
        rs += __shfl_xor(rs, 4, 32);
        rs += __shfl_xor(rs, 8, 32);
        lrun[r] = lrun[r] * alpha + rs;
#pragma unroll
        for (int nf = 0; nf < 4; ++nf) ctx[nf][r] *= alpha;
      }
      // ---- P: C-layout -> LDS -> A-layout (per-wave private region) ----
      __bf16* pw = &Ps[(size_t)wid * 16 * 72];
#pragma unroll
      for (int nf = 0; nf < 4; ++nf)
#pragma unroll
        for (int r = 0; r < 8; ++r)
          pw[(lhalf * 8 + r) * 72 + nf * 16 + l16] = (__bf16)s[nf][r];
      asm volatile("s_wait_dscnt 0" : : : "memory");  // same-wave LDS RAW
      // ---- ctx += P @ V ----
#pragma unroll
      for (int kk = 0; kk < 2; ++kk) {
        const __bf16* pr = &Ps[(size_t)wid * 16 * 72 + (size_t)l16 * 72 + kk * 32];
        const v16bf pa = ld_frag(pr + lhalf * 8, pr + 16 + lhalf * 8);
#pragma unroll
        for (int nf = 0; nf < 4; ++nf) {
          const __bf16* vr = &Vts[(nf * 16 + l16) * 72 + kk * 32];
          ctx[nf] = wmma_bf16(pa, ld_frag(vr + lhalf * 8, vr + 16 + lhalf * 8), ctx[nf]);
        }
      }
    }
  }

  // ---- normalize and store ctx as bf16 into flat [B*S, 1024] ----
  float inv[8];
#pragma unroll
  for (int r = 0; r < 8; ++r) inv[r] = 1.0f / lrun[r];
  const int b = bh >> 4, h = bh & 15;
#pragma unroll
  for (int nf = 0; nf < 4; ++nf) {
    const int col = h * 64 + nf * 16 + l16;
#pragma unroll
    for (int r = 0; r < 8; ++r) {
      const int m = b * S_LEN + qrow + lhalf * 8 + r;
      Ctx[(size_t)m * 1024 + col] = (__bf16)(ctx[nf][r] * inv[r]);
    }
  }
}

// ---------------------------------------------------------------------------
extern "C" void kernel_launch(void* const* d_in, const int* in_sizes, int n_in,
                              void* d_out, int out_size, void* d_ws, size_t ws_size,
                              hipStream_t stream) {
  const float* q  = (const float*)d_in[0];
  const float* k  = (const float*)d_in[1];
  const float* v  = (const float*)d_in[2];
  // d_in[3] = mask (causal, applied analytically in the kernel)
  const float* wq = (const float*)d_in[4];
  const float* bq = (const float*)d_in[5];
  const float* wk = (const float*)d_in[6];
  const float* bk = (const float*)d_in[7];
  const float* wv = (const float*)d_in[8];
  const float* bv = (const float*)d_in[9];
  const float* wo = (const float*)d_in[10];
  const float* bo = (const float*)d_in[11];

  const size_t HELEMS = (size_t)2 * NHEAD * S_LEN * DKDIM;  // 4,194,304
  __bf16* qh  = (__bf16*)d_ws;
  __bf16* kh  = qh + HELEMS;
  __bf16* vh  = kh + HELEMS;
  __bf16* ctx = vh + HELEMS;  // [4096 x 1024] bf16

  dim3 gproj(32, 8, 3);  // M/128, N/128, {q,k,v}
  qkv_proj_kernel<<<gproj, 256, 0, stream>>>(q, k, v, wq, bq, wk, bk, wv, bv,
                                             qh, kh, vh);
  dim3 gattn(S_LEN / 128, 2 * NHEAD);  // q-tiles, b*h
  attn_kernel<<<gattn, 256, 0, stream>>>(qh, kh, vh, ctx);
  dim3 gout(32, 8);
  out_proj_kernel<<<gout, 8 * 32, 0, stream>>>(ctx, wo, bo, (float*)d_out);
}
